// ScaledDotProduct_16733192585786
// MI455X (gfx1250) — matrix operand order repaired and checked by reading, not compiled, not run on hardware
//
#include <hip/hip_runtime.h>
#include <hip/hip_bf16.h>

// ---------------------------------------------------------------------------
// Scaled dot-product attention, fused single pass per 16-query tile.
// bf16 WMMA (v_wmma_f32_16x16x32_bf16), fp32 softmax + accumulation.
// Outputs: attn_output [B,H,S,D] fp32, then attn_weights [B,H,S,S] fp32.
// ---------------------------------------------------------------------------

typedef __attribute__((ext_vector_type(16))) __bf16 v16bf;
typedef __attribute__((ext_vector_type(8)))  float  v8f;
typedef __attribute__((ext_vector_type(4)))  float  v4f;

#define BB     2
#define HH     16
#define SS     2048
#define DD     128
#define NWAVES 16
#define NTHR   (NWAVES * 32)          // 512
#define KPW    (SS / NWAVES)          // 128 keys per wave
#define NT     (KPW / 16)             // 8 key tiles per wave
#define NDC    (DD / 32)              // 4 k-chunks for GEMM1 (K=32 each)
#define NDN    (DD / 16)              // 8 n-chunks for GEMM2
#define SCALE  0.08838834764831843f   // 1/sqrt(128)

__global__ __launch_bounds__(NTHR, 1)
void sdpa_fused_kernel(const float* __restrict__ Q,
                       const float* __restrict__ Kin,
                       const float* __restrict__ V,
                       float* __restrict__ Out,
                       float* __restrict__ Wout) {
    __shared__ __align__(16) __bf16 Qs[16 * DD];            // 4 KB
    __shared__ __align__(16) __bf16 Ps[NWAVES][16 * KPW];   // 64 KB
    __shared__ __align__(16) float  Os[16 * DD];            // 8 KB
    __shared__ float RedMax[16][NWAVES];                    // 1 KB
    __shared__ float RedSum[16][NWAVES];                    // 1 KB

    const int tid  = threadIdx.x;
    const int w    = tid >> 5;          // wave id 0..15
    const int lane = tid & 31;
    const int nIdx = lane & 15;         // tile column / matrix row (M) index
    const int half = lane >> 4;         // 0 or 1
    const int bh   = blockIdx.y;        // head index 0..31
    const int q0   = blockIdx.x * 16;   // query row base

    const float* Qh = Q   + (size_t)bh * SS * DD;
    const float* Kh = Kin + (size_t)bh * SS * DD;
    const float* Vh = V   + (size_t)bh * SS * DD;

    // ---- Phase 0: stage Q tile as bf16; zero output accumulator -----------
    {
        const int i4 = tid * 4;                 // 2048 floats / 512 thr = 4 each
        const int m = i4 >> 7, d = i4 & 127;
        v4f q = *reinterpret_cast<const v4f*>(Qh + (size_t)(q0 + m) * DD + d);
        Qs[i4 + 0] = (__bf16)q.x;  Qs[i4 + 1] = (__bf16)q.y;
        Qs[i4 + 2] = (__bf16)q.z;  Qs[i4 + 3] = (__bf16)q.w;
        *reinterpret_cast<v4f*>(&Os[i4]) = (v4f)0.0f;
    }
    __syncthreads();

    // ---- Phase 1: scores = (Q K^T) * scale --------------------------------
    // A operand (16x32) in this code's contiguous k order (B matches): lane l holds row M = l&15;
// element e maps to K = (l>>4)*16 + e  -> one
// contiguous 16-element run in LDS.
    v16bf a[NDC];
    #pragma unroll
    for (int dc = 0; dc < NDC; ++dc) {
        const int base0 = nIdx * DD + dc * 32 + half * 16;
        #pragma unroll
        for (int e = 0; e < 8; ++e) {
            a[dc][e]     = Qs[base0 + e];
            a[dc][e + 8] = Qs[base0 + 8 + e];
        }
    }

    v8f acc[NT];
    #pragma unroll
    for (int t = 0; t < NT; ++t) {
        acc[t] = (v8f)0.0f;
        const int kb = w * KPW + t * 16;
        #pragma unroll
        for (int dc = 0; dc < NDC; ++dc) {
            // B operand (32x16 = K^T chunk): lane holds key kb+nIdx,
            // rows d = dc*32 + half*16 + e  -> 16 contiguous floats.
            const float* kp = Kh + (size_t)(kb + nIdx) * DD + dc * 32 + half * 16;
            v16bf b;
            #pragma unroll
            for (int j = 0; j < 4; ++j) {
                v4f x = *reinterpret_cast<const v4f*>(kp + 4 * j);
                b[4 * j + 0] = (__bf16)x.x;  b[4 * j + 1] = (__bf16)x.y;
                b[4 * j + 2] = (__bf16)x.z;  b[4 * j + 3] = (__bf16)x.w;
            }
            acc[t] = __builtin_amdgcn_wmma_f32_16x16x32_bf16(
                false, a[dc], false, b, (short)0, acc[t], false, false);
        }
        #pragma unroll
        for (int g = 0; g < 8; ++g) acc[t][g] *= SCALE;
    }

    // ---- Phase 2: softmax over the full 2048-wide row ----------------------
    // C layout: VGPR g, lane -> row m = g + 8*half, col n = lane&15.
    float lred[8];
    #pragma unroll
    for (int g = 0; g < 8; ++g) {
        float v = -3.0e38f;
        #pragma unroll
        for (int t = 0; t < NT; ++t) v = fmaxf(v, acc[t][g]);
        #pragma unroll
        for (int sh = 1; sh < 16; sh <<= 1) v = fmaxf(v, __shfl_xor(v, sh, 32));
        lred[g] = v;
    }
    if (nIdx == 0) {
        #pragma unroll
        for (int g = 0; g < 8; ++g) RedMax[g + 8 * half][w] = lred[g];
    }
    __syncthreads();
    float gmax[8];
    #pragma unroll
    for (int g = 0; g < 8; ++g) {
        float v = -3.0e38f;
        for (int ww = 0; ww < NWAVES; ++ww) v = fmaxf(v, RedMax[g + 8 * half][ww]);
        gmax[g] = v;
    }

    #pragma unroll
    for (int g = 0; g < 8; ++g) lred[g] = 0.0f;
    #pragma unroll
    for (int t = 0; t < NT; ++t) {
        #pragma unroll
        for (int g = 0; g < 8; ++g) {
            float e = __expf(acc[t][g] - gmax[g]);
            acc[t][g] = e;
            lred[g] += e;
        }
    }
    #pragma unroll
    for (int g = 0; g < 8; ++g) {
        float v = lred[g];
        #pragma unroll
        for (int sh = 1; sh < 16; sh <<= 1) v += __shfl_xor(v, sh, 32);
        lred[g] = v;
    }
    if (nIdx == 0) {
        #pragma unroll
        for (int g = 0; g < 8; ++g) RedSum[g + 8 * half][w] = lred[g];
    }
    __syncthreads();
    float ginv[8];
    #pragma unroll
    for (int g = 0; g < 8; ++g) {
        float v = 0.0f;
        for (int ww = 0; ww < NWAVES; ++ww) v += RedSum[g + 8 * half][ww];
        ginv[g] = 1.0f / v;
    }

    // ---- Phase 2b: normalize; emit fp32 weights; stage bf16 P --------------
    #pragma unroll
    for (int t = 0; t < NT; ++t) {
        const int kb = w * KPW + t * 16;
        #pragma unroll
        for (int g = 0; g < 8; ++g) {
            const int m = g + 8 * half;
            const float p = acc[t][g] * ginv[g];
            Wout[((size_t)bh * SS + (q0 + m)) * SS + kb + nIdx] = p;
            Ps[w][m * KPW + t * 16 + nIdx] = (__bf16)p;   // same-wave use below
        }
    }

    // ---- Phase 3: partial Out = P * V over this wave's 128 keys ------------
    v8f oacc[NDN];
    #pragma unroll
    for (int dn = 0; dn < NDN; ++dn) oacc[dn] = (v8f)0.0f;

    #pragma unroll
    for (int kc = 0; kc < KPW / 32; ++kc) {          // 4 chunks of 32 keys
        v16bf pa;                                     // A operand from Ps
        const int base0 = nIdx * KPW + kc * 32 + half * 16;
        #pragma unroll
        for (int e = 0; e < 8; ++e) {
            pa[e]     = Ps[w][base0 + e];
            pa[e + 8] = Ps[w][base0 + 8 + e];
        }
        #pragma unroll
        for (int dn = 0; dn < NDN; ++dn) {
            // B operand (32x16 of V): lane holds col d = dn*16 + nIdx,
            // rows key = w*KPW + kc*32 + half*16 + e (stride DD floats).
            const float* vp = Vh + (size_t)(w * KPW + kc * 32 + half * 16) * DD
                                 + dn * 16 + nIdx;
            v16bf vb;
            #pragma unroll
            for (int e = 0; e < 16; ++e) vb[e] = (__bf16)vp[(size_t)e * DD];
            oacc[dn] = __builtin_amdgcn_wmma_f32_16x16x32_bf16(
                false, pa, false, vb, (short)0, oacc[dn], false, false);
        }
    }

    // ---- Phase 4: cross-wave reduce via LDS f32 atomics, then store --------
    #pragma unroll
    for (int dn = 0; dn < NDN; ++dn) {
        #pragma unroll
        for (int g = 0; g < 8; ++g) {
            const int m = g + 8 * half;
            atomicAdd(&Os[m * DD + dn * 16 + nIdx], oacc[dn][g]);
        }
    }
    __syncthreads();
    {
        const int i4 = tid * 4;
        const int m = i4 >> 7, d = i4 & 127;
        v4f o = *reinterpret_cast<const v4f*>(&Os[i4]);
        *reinterpret_cast<v4f*>(Out + ((size_t)bh * SS + (q0 + m)) * DD + d) = o;
    }
}

extern "C" void kernel_launch(void* const* d_in, const int* in_sizes, int n_in,
                              void* d_out, int out_size, void* d_ws, size_t ws_size,
                              hipStream_t stream) {
    (void)in_sizes; (void)n_in; (void)out_size; (void)d_ws; (void)ws_size;
    const float* q = (const float*)d_in[0];
    const float* k = (const float*)d_in[1];
    const float* v = (const float*)d_in[2];
    float* out  = (float*)d_out;                               // [B,H,S,D]
    float* wout = out + (size_t)BB * HH * SS * DD;             // [B,H,S,S]

    dim3 grid(SS / 16, BB * HH);   // (128, 32)
    dim3 block(NTHR);              // 512 threads = 16 wave32
    sdpa_fused_kernel<<<grid, block, 0, stream>>>(q, k, v, out, wout);
}
